// GQSWAttention_30262339568338
// MI455X (gfx1250) — compile-verified, matrix operand order checked by reading, not run on hardware
//
#include <hip/hip_runtime.h>
#include <hip/hip_bf16.h>
#include <stdint.h>

// ---------------------------------------------------------------------------
// GQA sliding-window attention for MI455X (gfx1250, wave32, WMMA bf16)
//   q = x@wq^T ; k = x@wk^T ; v = x@wv^T        (bf16 WMMA GEMMs, f32 accum)
//   RoPE(q, k)
//   banded attention (window=128, GQA 16q/4kv)   (bf16 WMMA, f32 softmax)
//   y = o@wo^T                                   (bf16 WMMA GEMM, f32 out)
// ---------------------------------------------------------------------------

typedef __attribute__((ext_vector_type(16))) __bf16 v16bf;
typedef __attribute__((ext_vector_type(8)))  __bf16 v8bf;
typedef __attribute__((ext_vector_type(8)))  float  v8f;

union Frag16 { v16bf v; v8bf h[2]; };
union Acc8   { v8f v; float f[8]; };
union B8     { v8bf v; __bf16 e[8]; };

__device__ __forceinline__ __bf16 f2bf(float f) { return (__bf16)f; }

__device__ __forceinline__ v8bf cvt8(const float* p) {
  const float4* q = (const float4*)p;
  float4 a = q[0], b = q[1];
  B8 r;
  r.e[0] = f2bf(a.x); r.e[1] = f2bf(a.y); r.e[2] = f2bf(a.z); r.e[3] = f2bf(a.w);
  r.e[4] = f2bf(b.x); r.e[5] = f2bf(b.y); r.e[6] = f2bf(b.z); r.e[7] = f2bf(b.w);
  return r.v;
}

// A/B fragment: lane m holds row m; halves at k = 8h..8h+7 and 16+8h..16+8h+7.
__device__ __forceinline__ v16bf load_frag(const __bf16* p, int h) {
  Frag16 u;
  u.h[0] = *(const v8bf*)(p + 8 * h);
  u.h[1] = *(const v8bf*)(p + 16 + 8 * h);
  return u.v;
}

__device__ __forceinline__ v8f wmma_bf16(v16bf a, v16bf b, v8f c) {
  return __builtin_amdgcn_wmma_f32_16x16x32_bf16(false, a, false, b, (short)0, c,
                                                 false, false);
}

// ---------------------------------------------------------------------------
// GEMM: C[M,N] = A[M,K] @ W[N,K]^T. Block tile 128x128, BK=32, 256 threads
// (8 waves). Wave (wm,wn) owns a 64x32 tile = 4x2 accumulators.
// ---------------------------------------------------------------------------
#define LDT 40  // LDS row stride in bf16 elems (80B = 20 banks, conflict-free)

template <bool A_BF16, bool C_BF16>
__global__ __launch_bounds__(256) void gemm_xwT(const void* __restrict__ Ap,
                                                const float* __restrict__ W,
                                                void* __restrict__ Cp,
                                                int M, int N, int K) {
  __shared__ __align__(16) __bf16 As[128 * LDT];
  __shared__ __align__(16) __bf16 Ws[128 * LDT];
  const int tid  = threadIdx.x;
  const int lane = tid & 31;
  const int wave = tid >> 5;
  const int wm   = wave >> 2;       // 0..1
  const int wn   = wave & 3;        // 0..3
  const int m16  = lane & 15;
  const int h    = lane >> 4;
  const int row0 = blockIdx.y * 128;
  const int col0 = blockIdx.x * 128;
  const int lr   = tid >> 1;        // 0..127  (cooperative-load row)
  const int lc   = (tid & 1) * 16;  // 0 / 16  (cooperative-load k offset)

  Acc8 acc[4][2];
#pragma unroll
  for (int i = 0; i < 4; ++i)
#pragma unroll
    for (int j = 0; j < 2; ++j)
#pragma unroll
      for (int e = 0; e < 8; ++e) acc[i][j].f[e] = 0.0f;

  const float*  Af32 = (const float*)Ap;
  const __bf16* Abf  = (const __bf16*)Ap;

  for (int kt = 0; kt < K; kt += 32) {
    if (A_BF16) {
      const __bf16* s = Abf + (size_t)(row0 + lr) * K + kt + lc;
      *(v8bf*)&As[lr * LDT + lc]     = *(const v8bf*)s;
      *(v8bf*)&As[lr * LDT + lc + 8] = *(const v8bf*)(s + 8);
    } else {
      const float* s = Af32 + (size_t)(row0 + lr) * K + kt + lc;
      *(v8bf*)&As[lr * LDT + lc]     = cvt8(s);
      *(v8bf*)&As[lr * LDT + lc + 8] = cvt8(s + 8);
    }
    {
      const float* s = W + (size_t)(col0 + lr) * K + kt + lc;
      *(v8bf*)&Ws[lr * LDT + lc]     = cvt8(s);
      *(v8bf*)&Ws[lr * LDT + lc + 8] = cvt8(s + 8);
    }
    __syncthreads();

    v16bf af[4], bfr[2];
#pragma unroll
    for (int i = 0; i < 4; ++i)
      af[i] = load_frag(&As[(wm * 64 + i * 16 + m16) * LDT], h);
#pragma unroll
    for (int j = 0; j < 2; ++j)
      bfr[j] = load_frag(&Ws[(wn * 32 + j * 16 + m16) * LDT], h);
#pragma unroll
    for (int i = 0; i < 4; ++i)
#pragma unroll
      for (int j = 0; j < 2; ++j)
        acc[i][j].v = wmma_bf16(af[i], bfr[j], acc[i][j].v);
    __syncthreads();
  }

#pragma unroll
  for (int i = 0; i < 4; ++i)
#pragma unroll
    for (int j = 0; j < 2; ++j)
#pragma unroll
      for (int e = 0; e < 8; ++e) {
        int r = row0 + wm * 64 + i * 16 + e + 8 * h;
        int c = col0 + wn * 32 + j * 16 + m16;
        if (C_BF16)
          ((__bf16*)Cp)[(size_t)r * N + c] = f2bf(acc[i][j].f[e]);
        else
          ((float*)Cp)[(size_t)r * N + c] = acc[i][j].f[e];
      }
}

// ---------------------------------------------------------------------------
// RoPE (rotate-half), in place on bf16, Hc heads of dim 128, T=4096.
// ---------------------------------------------------------------------------
__global__ __launch_bounds__(256) void rope_kernel(__bf16* p, int Hc,
                                                   const int* pos0) {
  size_t gid = (size_t)blockIdx.x * 256 + threadIdx.x;
  int d      = (int)(gid & 63);
  size_t hv  = gid >> 6;
  int t      = (int)((hv / (size_t)Hc) % 4096);
  size_t base = hv * 128;
  float pos  = (float)(pos0[0] + t);
  // 10000^(-2d/128) = exp(-d * ln(10000)/64)
  float inv  = __expf((float)d * -0.14391156831212787f);
  float ang  = pos * inv;
  float c, s;
  __sincosf(ang, &s, &c);
  float x1 = (float)p[base + d];
  float x2 = (float)p[base + d + 64];
  p[base + d]      = f2bf(x1 * c - x2 * s);
  p[base + d + 64] = f2bf(x2 * c + x1 * s);
}

// ---------------------------------------------------------------------------
// Banded attention. One workgroup per (b, block n, kv head, group head g).
// 8 waves; wave w owns 16 query rows [16w, 16w+16). Keys 0..255 map to
// positions (n-1)*128 + j.  valid(i,j) = (i+1 <= j <= i+128) [& j>=128 if n==0].
// Chunk loops are FULLY UNROLLED so all accumulators use constant indices and
// stay in VGPRs (no scratch spills). Per-wave activity is a scalar branch.
// ---------------------------------------------------------------------------
__global__ __launch_bounds__(256) void attn_kernel(const __bf16* __restrict__ q,
                                                   const __bf16* __restrict__ k,
                                                   const __bf16* __restrict__ v,
                                                   __bf16* __restrict__ o) {
  constexpr int T = 4096, NH = 16, KVH = 4, GRP = 4, HD = 128;
  __shared__ __align__(16) __bf16 kv_lds[128 * 40];  // k chunk (32x136) / v^T (128x40)
  __shared__ __align__(16) __bf16 p_lds[8 * 16 * 40];

  const int bid  = blockIdx.x;
  const int g    = bid & 3;
  const int kvh  = (bid >> 2) & 3;
  const int n    = (bid >> 4) & 31;
  const int b    = bid >> 9;
  const int head = kvh * GRP + g;

  const int tid  = threadIdx.x;
  const int lane = tid & 31;
  const int w    = tid >> 5;   // wave 0..7
  const int m16  = lane & 15;
  const int h    = lane >> 4;
  const int r0   = 16 * w;     // wave's first query row within the block

  // ---- resident q fragments (16 rows x 128 d = 4 K-chunks) ----
  const __bf16* qrow =
      q + ((size_t)(b * T + n * 128 + r0 + m16) * NH + head) * HD;
  v16bf qf[4];
#pragma unroll
  for (int kc = 0; kc < 4; ++kc) qf[kc] = load_frag(qrow + kc * 32, h);

  Acc8 sacc[16];
#pragma unroll
  for (int idx = 0; idx < 16; ++idx)
#pragma unroll
    for (int e = 0; e < 8; ++e) sacc[idx].f[e] = 0.0f;

  const int jc0 = (n == 0) ? 4 : 0;  // block-uniform

  // cooperative-load lane mapping (shared by both passes)
  const int cj  = tid >> 3;        // key row within chunk 0..31
  const int cd0 = (tid & 7) * 16;  // d offset 0..112

  // ---------------- QK^T pass ----------------
#pragma unroll
  for (int jc = 0; jc < 8; ++jc) {
    if (jc >= jc0) {  // block-uniform; load 32 keys x 128 d -> LDS stride 136
      int tk = n * 128 - 128 + jc * 32 + cj;
      const __bf16* kr = k + ((size_t)(b * T + tk) * KVH + kvh) * HD + cd0;
      *(v8bf*)&kv_lds[cj * 136 + cd0]     = *(const v8bf*)kr;
      *(v8bf*)&kv_lds[cj * 136 + cd0 + 8] = *(const v8bf*)(kr + 8);
    }
    __syncthreads();
    // wave-uniform activity -> scalar branch
    int act = __builtin_amdgcn_readfirstlane(
        (jc >= jc0 && 32 * jc + 31 >= r0 + 1 && 32 * jc <= r0 + 143) ? 1 : 0);
    if (act) {
      v16bf kf[8];
#pragma unroll
      for (int tn = 0; tn < 2; ++tn)
#pragma unroll
        for (int kc = 0; kc < 4; ++kc)
          kf[tn * 4 + kc] =
              load_frag(&kv_lds[(tn * 16 + m16) * 136 + kc * 32], h);
#pragma unroll
      for (int tn = 0; tn < 2; ++tn)
#pragma unroll
        for (int kc = 0; kc < 4; ++kc)
          sacc[jc * 2 + tn].v =
              wmma_bf16(qf[kc], kf[tn * 4 + kc], sacc[jc * 2 + tn].v);
    }
    __syncthreads();
  }

  // ---------------- mask + softmax (in C-layout registers) ----------------
  const float scale = 0.08838834764831845f;  // 1/sqrt(128)
  float rm[8], rs[8];
#pragma unroll
  for (int e = 0; e < 8; ++e) rm[e] = -3.0e38f;
#pragma unroll
  for (int idx = 0; idx < 16; ++idx) {
    int jc = idx >> 1, tn = idx & 1;
    int j = jc * 32 + tn * 16 + m16;
#pragma unroll
    for (int e = 0; e < 8; ++e) {
      int i = r0 + e + 8 * h;
      bool valid = (j >= i + 1) && (j <= i + 128) && (n > 0 || j >= 128);
      float s = valid ? sacc[idx].f[e] * scale : -3.0e38f;
      sacc[idx].f[e] = s;
      rm[e] = fmaxf(rm[e], s);
    }
  }
#pragma unroll
  for (int e = 0; e < 8; ++e) {
    rm[e] = fmaxf(rm[e], __shfl_xor(rm[e], 1));
    rm[e] = fmaxf(rm[e], __shfl_xor(rm[e], 2));
    rm[e] = fmaxf(rm[e], __shfl_xor(rm[e], 4));
    rm[e] = fmaxf(rm[e], __shfl_xor(rm[e], 8));
    rs[e] = 0.0f;
  }
#pragma unroll
  for (int idx = 0; idx < 16; ++idx)
#pragma unroll
    for (int e = 0; e < 8; ++e) {
      float ex = __expf(sacc[idx].f[e] - rm[e]);
      sacc[idx].f[e] = ex;
      rs[e] += ex;
    }
#pragma unroll
  for (int e = 0; e < 8; ++e) {
    rs[e] += __shfl_xor(rs[e], 1);
    rs[e] += __shfl_xor(rs[e], 2);
    rs[e] += __shfl_xor(rs[e], 4);
    rs[e] += __shfl_xor(rs[e], 8);
    rs[e] = 1.0f / rs[e];
  }
#pragma unroll
  for (int idx = 0; idx < 16; ++idx)
#pragma unroll
    for (int e = 0; e < 8; ++e) sacc[idx].f[e] *= rs[e];

  // ---------------- PV pass ----------------
  Acc8 oacc[8];
#pragma unroll
  for (int td = 0; td < 8; ++td)
#pragma unroll
    for (int e = 0; e < 8; ++e) oacc[td].f[e] = 0.0f;

#pragma unroll
  for (int jc = 0; jc < 8; ++jc) {
    int act = __builtin_amdgcn_readfirstlane(
        (jc >= jc0 && 32 * jc + 31 >= r0 + 1 && 32 * jc <= r0 + 143) ? 1 : 0);
    if (jc >= jc0) {  // load v chunk transposed: vt[d*40 + j]
      int tk = n * 128 - 128 + jc * 32 + cj;
      const __bf16* vr = v + ((size_t)(b * T + tk) * KVH + kvh) * HD + cd0;
      B8 a, c;
      a.v = *(const v8bf*)vr;
      c.v = *(const v8bf*)(vr + 8);
#pragma unroll
      for (int e = 0; e < 8; ++e) kv_lds[(cd0 + e) * 40 + cj] = a.e[e];
#pragma unroll
      for (int e = 0; e < 8; ++e) kv_lds[(cd0 + 8 + e) * 40 + cj] = c.e[e];
    }
    if (act) {  // stage this wave's normalized p chunk (16x32) as bf16
#pragma unroll
      for (int tn = 0; tn < 2; ++tn)
#pragma unroll
        for (int e = 0; e < 8; ++e)
          p_lds[w * 640 + (e + 8 * h) * 40 + tn * 16 + m16] =
              f2bf(sacc[jc * 2 + tn].f[e]);
    }
    __syncthreads();
    if (act) {
      v16bf pf = load_frag(&p_lds[w * 640 + m16 * 40], h);
      v16bf vf[8];
#pragma unroll
      for (int td = 0; td < 8; ++td)
        vf[td] = load_frag(&kv_lds[(td * 16 + m16) * 40], h);
#pragma unroll
      for (int td = 0; td < 8; ++td)
        oacc[td].v = wmma_bf16(pf, vf[td], oacc[td].v);
    }
    __syncthreads();
  }

  // ---------------- write o (bf16), layout (b, t, head*128 + d) ----------------
  const int tq = n * 128 + r0;
#pragma unroll
  for (int td = 0; td < 8; ++td)
#pragma unroll
    for (int e = 0; e < 8; ++e) {
      size_t r = (size_t)(b * T + tq + e + 8 * h);
      o[r * 2048 + head * 128 + td * 16 + m16] = f2bf(oacc[td].f[e]);
    }
}

// ---------------------------------------------------------------------------
// Host launcher
// ---------------------------------------------------------------------------
extern "C" void kernel_launch(void* const* d_in, const int* in_sizes, int n_in,
                              void* d_out, int out_size, void* d_ws,
                              size_t ws_size, hipStream_t stream) {
  (void)in_sizes; (void)n_in; (void)out_size; (void)ws_size;
  const float* x  = (const float*)d_in[0];
  const float* wq = (const float*)d_in[1];
  const float* wk = (const float*)d_in[2];
  const float* wv = (const float*)d_in[3];
  const float* wo = (const float*)d_in[4];
  const int* pos0 = (const int*)d_in[5];
  float* out = (float*)d_out;

  const int B = 2, T = 4096, DIM = 2048, KVD = 512;
  const size_t M = (size_t)B * T;  // 8192

  __bf16* qb = (__bf16*)d_ws;                  // 8192*2048
  __bf16* kb = qb + M * DIM;                   // 8192*512
  __bf16* vb = kb + M * KVD;                   // 8192*512
  __bf16* ob = vb + M * KVD;                   // 8192*2048

  dim3 blk(256);
  // Projections: q, k, v (f32 in -> bf16 out)
  gemm_xwT<false, true><<<dim3(DIM / 128, M / 128), blk, 0, stream>>>(
      x, wq, qb, (int)M, DIM, DIM);
  gemm_xwT<false, true><<<dim3(KVD / 128, M / 128), blk, 0, stream>>>(
      x, wk, kb, (int)M, KVD, DIM);
  gemm_xwT<false, true><<<dim3(KVD / 128, M / 128), blk, 0, stream>>>(
      x, wv, vb, (int)M, KVD, DIM);

  // RoPE on q (16 heads) and k (4 heads)
  rope_kernel<<<(unsigned)(M * 16 * 64 / 256), blk, 0, stream>>>(qb, 16, pos0);
  rope_kernel<<<(unsigned)(M * 4 * 64 / 256), blk, 0, stream>>>(kb, 4, pos0);

  // Attention: grid = B * nb * KV * G = 2*32*4*4 = 1024
  attn_kernel<<<1024, blk, 0, stream>>>(qb, kb, vb, ob);

  // Output projection (bf16 in -> f32 out)
  gemm_xwT<true, false><<<dim3(DIM / 128, M / 128), blk, 0, stream>>>(
      ob, wo, out, (int)M, DIM, DIM);
}
